// ConsciousnessAwareAttention_56487409877284
// MI455X (gfx1250) — matrix operand order repaired and checked by reading, not compiled, not run on hardware
//
#include <hip/hip_runtime.h>

#define DM 512
#define NH 8
#define DK 64
#define BB 2
#define SS 4096

typedef __attribute__((ext_vector_type(16))) _Float16 v16h;
typedef __attribute__((ext_vector_type(8)))  float    v8f;

static __device__ __forceinline__ v8f wmma16(v16h a, v16h b, v8f c) {
  return __builtin_amdgcn_wmma_f32_16x16x32_f16(false, a, false, b, (short)0, c,
                                                false, false);
}

// A-operand 16x32 f16 tile from a row-major f16 matrix.
// ISA layout: lanes 0-15 row M=lane, elems 0..7 = K 8*hf.., elems 8..15 = K 16+8*hf..
static __device__ __forceinline__ v16h load_a16(const _Float16* p, int ld,
                                                int row0, int k0, int lane) {
  const int ll = lane & 15, hf = lane >> 4;
  const _Float16* q = p + (size_t)(row0 + ll) * ld + k0;
  v16h a;
#pragma unroll
  for (int i = 0; i < 8; ++i) a[i] = q[8 * hf + i];
#pragma unroll
  for (int i = 0; i < 8; ++i) a[8 + i] = q[16 + 8 * hf + i];
  return a;
}

// Same but converting from f32 source on the fly.
static __device__ __forceinline__ v16h load_a32(const float* p, int ld,
                                                int row0, int k0, int lane) {
  const int ll = lane & 15, hf = lane >> 4;
  const float* q = p + (size_t)(row0 + ll) * ld + k0;
  v16h a;
#pragma unroll
  for (int i = 0; i < 8; ++i) a[i] = (_Float16)q[8 * hf + i];
#pragma unroll
  for (int i = 0; i < 8; ++i) a[8 + i] = (_Float16)q[16 + 8 * hf + i];
  return a;
}

// B-operand 32x16 f16 tile where the source is stored N-major: B[k][n] = p[n*ld + k].
// Per-lane access is two runs of 8 contiguous halves (16 bytes) -> vectorizable loads.
// ISA layout: N = lane&15, elems 0..7 = K 8*hf.., elems 8..15 = K 16+8*hf..
static __device__ __forceinline__ v16h load_bt(const _Float16* p, int ld,
                                               int k0, int n0, int lane) {
  const int ll = lane & 15, hf = lane >> 4;
  const _Float16* q = p + (size_t)(n0 + ll) * ld + k0;
  v16h b;
#pragma unroll
  for (int i = 0; i < 8; ++i) b[i] = q[8 * hf + i];
#pragma unroll
  for (int i = 0; i < 8; ++i) b[8 + i] = q[16 + 8 * hf + i];
  return b;
}

// ---------------------------------------------------------------------------
// Weight convert + transpose: w[k][n] f32 -> wt[n][k] f16  (512x512)
__global__ void wcvt_kernel(const float* __restrict__ w, _Float16* __restrict__ wt) {
  const int idx = blockIdx.x * 256 + threadIdx.x;   // over 512*512
  const int k = idx >> 9, n = idx & (DM - 1);
  wt[(size_t)n * DM + k] = (_Float16)w[idx];
}

// ---------------------------------------------------------------------------
// Gates: cw[b][h] = sigmoid(cs@w_gate + b_gate); iw[b][c] = sigmoid(cs@w_int + b_int)
__global__ void gates_kernel(const float* __restrict__ cs,
                             const float* __restrict__ wg, const float* __restrict__ bg,
                             const float* __restrict__ wi, const float* __restrict__ bi,
                             float* __restrict__ cw, float* __restrict__ iw) {
  const int t = blockIdx.x * blockDim.x + threadIdx.x;
  if (t < BB * NH) {
    const int b = t >> 3, h = t & 7;
    float acc = bg[h];
    for (int k = 0; k < DM; ++k) acc += cs[b * DM + k] * wg[k * NH + h];
    cw[t] = 1.0f / (1.0f + __expf(-acc));
  }
  if (t < BB * DM) {
    const int b = t >> 9, c = t & (DM - 1);
    float acc = bi[c];
    for (int k = 0; k < DM; ++k) acc += cs[b * DM + k] * wi[k * DM + c];
    iw[t] = 1.0f / (1.0f + __expf(-acc));
  }
}

// ---------------------------------------------------------------------------
// QKV projection GEMM: x[8192x512] @ {wq,wk,wv}[512x512] -> f16 head-major buffers.
// Q,K stored [b][h][s][dk]; V stored transposed [b][h][dk][s] so the attn kernel's
// B-operand loads are contiguous. One wave per 16x16 output tile, 8 waves/block.
__global__ __launch_bounds__(256)
void qkv_kernel(const float* __restrict__ x,
                const _Float16* __restrict__ wqT, const _Float16* __restrict__ wkT,
                const _Float16* __restrict__ wvT,
                _Float16* __restrict__ qb, _Float16* __restrict__ kb,
                _Float16* __restrict__ vb) {
  const int lane = threadIdx.x & 31;
  const int wave = threadIdx.x >> 5;
  const int ll = lane & 15, hf = lane >> 4;
  const int m0 = blockIdx.x * 16;
  const int n0 = (blockIdx.y * 8 + wave) * 16;
  v8f cq = {}, ck = {}, cv = {};
  for (int k0 = 0; k0 < DM; k0 += 32) {
    v16h a  = load_a32(x, DM, m0, k0, lane);
    v16h bq = load_bt(wqT, DM, k0, n0, lane);
    v16h bk = load_bt(wkT, DM, k0, n0, lane);
    v16h bv = load_bt(wvT, DM, k0, n0, lane);
    cq = wmma16(a, bq, cq);
    ck = wmma16(a, bk, ck);
    cv = wmma16(a, bv, cv);
  }
  const int c = n0 + ll;         // output feature (tile lies inside one head)
  const int h = c >> 6;
  const int dk = c & (DK - 1);
#pragma unroll
  for (int r = 0; r < 8; ++r) {
    const int m = m0 + 8 * hf + r;      // global row over B*S
    const int bi = m >> 12;             // S = 4096
    const int s  = m & (SS - 1);
    const size_t base = (size_t)(bi * NH + h);
    qb[(base * SS + s) * DK + dk] = (_Float16)cq[r];
    kb[(base * SS + s) * DK + dk] = (_Float16)ck[r];
    vb[(base * DK + dk) * SS + s] = (_Float16)cv[r];  // transposed V
  }
}

// ---------------------------------------------------------------------------
// Fused attention: one block per (batch, 16-query tile); wave w = head w.
// Pass A: streaming softmax stats (online max / sumexp) over all keys.
// Pass B: recompute scores, emit probabilities, LDS-reduce head mean -> amean,
//         accumulate context via WMMA.  Then: output projection GEMM + bias +
//         integration gate + residual + LayerNorm.
__global__ __launch_bounds__(256)
void attn_kernel(const _Float16* __restrict__ qb, const _Float16* __restrict__ kb,
                 const _Float16* __restrict__ vb, const float* __restrict__ x,
                 const _Float16* __restrict__ woT, const float* __restrict__ bo,
                 const float* __restrict__ cw, const float* __restrict__ iw,
                 const float* __restrict__ lng, const float* __restrict__ lnb,
                 float* __restrict__ outp, float* __restrict__ amean) {
  __shared__ float    fbuf[16 * DM];     // 32KB: pass B p-tiles [8][16][32]; later h-values [16][512]
  __shared__ _Float16 ctxbuf[16 * DM];   // 16KB: context tile [16][512] f16

  const int lane = threadIdx.x & 31;
  const int wave = threadIdx.x >> 5;
  const int ll = lane & 15, hf = lane >> 4;
  const int b  = blockIdx.x >> 8;                 // 256 q-tiles per batch
  const int q0 = (blockIdx.x & 255) * 16;
  const int h  = wave;

  const _Float16* Q = qb + (size_t)(b * NH + h) * SS * DK;   // [s][dk]
  const _Float16* K = kb + (size_t)(b * NH + h) * SS * DK;   // [s][dk]
  const _Float16* V = vb + (size_t)(b * NH + h) * DK * SS;   // [dk][s]
  const float scale = cw[b * NH + h] * 0.125f;               // cw / sqrt(64)

  const v16h qa0 = load_a16(Q, DK, q0, 0, lane);
  const v16h qa1 = load_a16(Q, DK, q0, 32, lane);

  // ---- Pass A: row stats (register r holds row 8*hf + r, cols = 16-lane group)
  float m[8], l[8];
#pragma unroll
  for (int r = 0; r < 8; ++r) { m[r] = -1e30f; l[r] = 0.0f; }

  for (int k0 = 0; k0 < SS; k0 += 16) {
    v16h t0 = load_bt(K, DK, 0, k0, lane);    // B[k=dk][n=key] = K[key][dk]
    v16h t1 = load_bt(K, DK, 32, k0, lane);
    v8f s = {};
    s = wmma16(qa0, t0, s);
    s = wmma16(qa1, t1, s);
#pragma unroll
    for (int r = 0; r < 8; ++r) {
      const float v = s[r] * scale;
      float mx = v;
      mx = fmaxf(mx, __shfl_xor(mx, 1));
      mx = fmaxf(mx, __shfl_xor(mx, 2));
      mx = fmaxf(mx, __shfl_xor(mx, 4));
      mx = fmaxf(mx, __shfl_xor(mx, 8));
      const float mn = fmaxf(m[r], mx);
      float e = __expf(v - mn);
      e += __shfl_xor(e, 1);
      e += __shfl_xor(e, 2);
      e += __shfl_xor(e, 4);
      e += __shfl_xor(e, 8);
      l[r] = l[r] * __expf(m[r] - mn) + e;
      m[r] = mn;
    }
  }
  float rl[8];
#pragma unroll
  for (int r = 0; r < 8; ++r) rl[r] = 1.0f / l[r];

  // ---- Pass B: probabilities + head-mean + context accumulation
  v8f o0 = {}, o1 = {}, o2 = {}, o3 = {};
  float* pbuf = fbuf + h * 512;                  // this head's 16x32 slot
  for (int k0 = 0; k0 < SS; k0 += 32) {
    v8f s0 = {}, s1 = {};
    {
      v16h t0 = load_bt(K, DK, 0, k0, lane);
      v16h t1 = load_bt(K, DK, 32, k0, lane);
      s0 = wmma16(qa0, t0, s0);
      s0 = wmma16(qa1, t1, s0);
    }
    {
      v16h t0 = load_bt(K, DK, 0, k0 + 16, lane);
      v16h t1 = load_bt(K, DK, 32, k0 + 16, lane);
      s1 = wmma16(qa0, t0, s1);
      s1 = wmma16(qa1, t1, s1);
    }
#pragma unroll
    for (int r = 0; r < 8; ++r) {
      const int row = 8 * hf + r;
      pbuf[row * 32 + ll]      = __expf(s0[r] * scale - m[r]) * rl[r];
      pbuf[row * 32 + 16 + ll] = __expf(s1[r] * scale - m[r]) * rl[r];
    }
    __syncthreads();
    // mean over heads -> attn_mean output (each element written exactly once)
    for (int e = threadIdx.x; e < 512; e += 256) {
      const int r = e >> 5, c = e & 31;
      float acc = 0.0f;
#pragma unroll
      for (int hh = 0; hh < NH; ++hh) acc += fbuf[hh * 512 + r * 32 + c];
      amean[(size_t)(b * SS + q0 + r) * SS + (k0 + c)] = acc * 0.125f;
    }
    // re-read own probabilities in A-operand layout (row = ll)
    v16h ap;
    {
      const float* pr = pbuf + ll * 32;
#pragma unroll
      for (int i = 0; i < 8; ++i) ap[i] = (_Float16)pr[8 * hf + i];
#pragma unroll
      for (int i = 0; i < 8; ++i) ap[8 + i] = (_Float16)pr[16 + 8 * hf + i];
    }
    v16h bv0 = load_bt(V, SS, k0, 0, lane);   // B[k=key][n=dk] = vb[dk][key]
    v16h bv1 = load_bt(V, SS, k0, 16, lane);
    v16h bv2 = load_bt(V, SS, k0, 32, lane);
    v16h bv3 = load_bt(V, SS, k0, 48, lane);
    o0 = wmma16(ap, bv0, o0);
    o1 = wmma16(ap, bv1, o1);
    o2 = wmma16(ap, bv2, o2);
    o3 = wmma16(ap, bv3, o3);
    __syncthreads();
  }

  // ---- context -> LDS [16][512] f16 (head h owns cols h*64 .. h*64+63)
#pragma unroll
  for (int r = 0; r < 8; ++r) {
    const int row = 8 * hf + r;
    _Float16* cr = ctxbuf + row * DM + h * DK + ll;
    cr[0]  = (_Float16)o0[r];
    cr[16] = (_Float16)o1[r];
    cr[32] = (_Float16)o2[r];
    cr[48] = (_Float16)o3[r];
  }
  __syncthreads();

  // ---- output projection: ctx[16x512] @ w_o[512x512]; wave w owns cols w*64..+63
  v8f a0 = {}, a1 = {}, a2 = {}, a3 = {};
  for (int k0 = 0; k0 < DM; k0 += 32) {
    v16h a  = load_a16(ctxbuf, DM, 0, k0, lane);
    v16h w0 = load_bt(woT, DM, k0, wave * 64 + 0, lane);
    v16h w1 = load_bt(woT, DM, k0, wave * 64 + 16, lane);
    v16h w2 = load_bt(woT, DM, k0, wave * 64 + 32, lane);
    v16h w3 = load_bt(woT, DM, k0, wave * 64 + 48, lane);
    a0 = wmma16(a, w0, a0);
    a1 = wmma16(a, w1, a1);
    a2 = wmma16(a, w2, a2);
    a3 = wmma16(a, w3, a3);
  }

  // ---- epilogue: bias + integration gate + residual -> fbuf (pbuf is dead)
#pragma unroll
  for (int j = 0; j < 4; ++j) {
    const v8f acc = (j == 0) ? a0 : (j == 1) ? a1 : (j == 2) ? a2 : a3;
    const int col = wave * 64 + j * 16 + ll;
    const float gv = iw[b * DM + col];
    const float bv = bo[col];
#pragma unroll
    for (int r = 0; r < 8; ++r) {
      const int row = 8 * hf + r;
      fbuf[row * DM + col] =
          (acc[r] + bv) * gv + x[(size_t)(b * SS + q0 + row) * DM + col];
    }
  }
  __syncthreads();

  // ---- LayerNorm: each wave normalizes 2 rows
#pragma unroll
  for (int rr = 0; rr < 2; ++rr) {
    const int row = wave * 2 + rr;
    const float* hr = fbuf + row * DM;
    float s1 = 0.0f, s2 = 0.0f;
    for (int c = lane; c < DM; c += 32) { const float v = hr[c]; s1 += v; s2 += v * v; }
#pragma unroll
    for (int off = 16; off > 0; off >>= 1) {
      s1 += __shfl_xor(s1, off);
      s2 += __shfl_xor(s2, off);
    }
    const float mu  = s1 * (1.0f / DM);
    const float var = s2 * (1.0f / DM) - mu * mu;
    const float inv = rsqrtf(var + 1e-5f);
    float* orow = outp + (size_t)(b * SS + q0 + row) * DM;
    for (int c = lane; c < DM; c += 32)
      orow[c] = (hr[c] - mu) * inv * lng[c] + lnb[c];
  }
}

// ---------------------------------------------------------------------------
extern "C" void kernel_launch(void* const* d_in, const int* in_sizes, int n_in,
                              void* d_out, int out_size, void* d_ws, size_t ws_size,
                              hipStream_t stream) {
  (void)in_sizes; (void)n_in; (void)out_size; (void)ws_size;
  const float* x   = (const float*)d_in[0];
  const float* cs  = (const float*)d_in[1];
  const float* wq  = (const float*)d_in[2];
  const float* wk  = (const float*)d_in[3];
  const float* wv  = (const float*)d_in[4];
  const float* wo  = (const float*)d_in[5];
  const float* bo  = (const float*)d_in[6];
  const float* wg  = (const float*)d_in[7];
  const float* bg  = (const float*)d_in[8];
  const float* wi  = (const float*)d_in[9];
  const float* bi  = (const float*)d_in[10];
  const float* lng = (const float*)d_in[11];
  const float* lnb = (const float*)d_in[12];

  float* outp  = (float*)d_out;                      // [B,S,D]
  float* amean = outp + (size_t)BB * SS * DM;        // [B,S,S]

  // workspace layout (~27.3 MB)
  const size_t NQKV = (size_t)BB * NH * SS * DK;     // 4,194,304 halves each
  _Float16* qb  = (_Float16*)d_ws;
  _Float16* kb  = qb + NQKV;
  _Float16* vb  = kb + NQKV;
  _Float16* wqT = vb + NQKV;
  _Float16* wkT = wqT + (size_t)DM * DM;
  _Float16* wvT = wkT + (size_t)DM * DM;
  _Float16* woT = wvT + (size_t)DM * DM;
  float* cwp = (float*)(woT + (size_t)DM * DM);
  float* iwp = cwp + BB * NH;

  const int wblocks = DM * DM / 256;                 // 1024
  wcvt_kernel<<<wblocks, 256, 0, stream>>>(wq, wqT);
  wcvt_kernel<<<wblocks, 256, 0, stream>>>(wk, wkT);
  wcvt_kernel<<<wblocks, 256, 0, stream>>>(wv, wvT);
  wcvt_kernel<<<wblocks, 256, 0, stream>>>(wo, woT);

  gates_kernel<<<4, 256, 0, stream>>>(cs, wg, bg, wi, bi, cwp, iwp);

  qkv_kernel<<<dim3((BB * SS) / 16, (DM / 16) / 8), 256, 0, stream>>>(
      x, wqT, wkT, wvT, qb, kb, vb);

  attn_kernel<<<BB * (SS / 16), 256, 0, stream>>>(
      qb, kb, vb, x, woT, bo, cwp, iwp, lng, lnb, outp, amean);
}
